// GCN_v2_5652176962022
// MI455X (gfx1250) — compile-verified, matrix operand order checked by reading, not run on hardware
//
#include <hip/hip_runtime.h>
#include <hip/hip_bf16.h>
#include <stdint.h>

#define NN 50000
#define EE 800000
#define HH 128
#define GG 250
#define EPS 1e-5f

typedef __attribute__((ext_vector_type(16))) __bf16        v16bf;
typedef __attribute__((ext_vector_type(8)))  float         v8f;
typedef __attribute__((ext_vector_type(4)))  unsigned int  v4u;
typedef __attribute__((ext_vector_type(8)))  int           v8i;
typedef __attribute__((ext_vector_type(4)))  int           v4i;

#if __has_builtin(__builtin_amdgcn_tensor_load_to_lds)
#define HAVE_TDM 1
#else
#define HAVE_TDM 0
#endif

// ---------------------------------------------------------------------------
// small utility kernels
// ---------------------------------------------------------------------------
__global__ void zero_f32(float* __restrict__ p, int n) {
    int i = blockIdx.x * blockDim.x + threadIdx.x;
    if (i < n) p[i] = 0.0f;
}

__global__ void deg_count(const int* __restrict__ dst, float* __restrict__ deg, int e) {
    int i = blockIdx.x * blockDim.x + threadIdx.x;
    if (i < e) unsafeAtomicAdd(&deg[dst[i]], 1.0f);
}

__global__ void deg_final(const float* __restrict__ deg, float* __restrict__ dis,
                          float* __restrict__ inv, int n) {
    int i = blockIdx.x * blockDim.x + threadIdx.x;
    if (i < n) {
        float d = deg[i] + 1.0f;          // +1 self loop
        dis[i]  = rsqrtf(d);
        inv[i]  = 1.0f / d;
    }
}

// out[n,f] = h[n,f] * inv_deg[n] + bias[f]     (also fully initializes `out`)
__global__ void self_init(const float* __restrict__ h, const float* __restrict__ inv,
                          const float* __restrict__ bias, float* __restrict__ out, int n) {
    int i = blockIdx.x * blockDim.x + threadIdx.x;
    if (i < n * HH) {
        int node = i >> 7, f = i & 127;
        out[i] = h[i] * inv[node] + bias[f];
    }
}

// scatter-add normalized messages: out[dst] += h[src] * dis[src]*dis[dst]
__global__ void __launch_bounds__(256)
edge_scatter(const float* __restrict__ h, const int* __restrict__ src,
             const int* __restrict__ dst, const float* __restrict__ dis,
             float* __restrict__ out, int e) {
    int eid = blockIdx.x * 2 + (threadIdx.x >> 7);
    int f   = threadIdx.x & 127;
    if (eid < e) {
        int s = src[eid], d = dst[eid];
        float c = dis[s] * dis[d];
        unsafeAtomicAdd(&out[(size_t)d * HH + f], h[(size_t)s * HH + f] * c);
    }
}

// per-feature sum / sum-of-squares (register partials, then atomics)
__global__ void __launch_bounds__(256)
bn_stats(const float* __restrict__ a, float* __restrict__ sum,
         float* __restrict__ sq, int n) {
    int f = threadIdx.x & 127;
    int r = blockIdx.x * 2 + (threadIdx.x >> 7);
    float s = 0.0f, q = 0.0f;
    for (; r < n; r += gridDim.x * 2) {
        float v = a[(size_t)r * HH + f];
        s += v; q += v * v;
    }
    unsafeAtomicAdd(&sum[f], s);
    unsafeAtomicAdd(&sq[f], q);
}

// fused BatchNorm (training stats) + ReLU, in place
__global__ void bn_apply(float* __restrict__ a, const float* __restrict__ sum,
                         const float* __restrict__ sq, const float* __restrict__ g,
                         const float* __restrict__ be, int n) {
    int i = blockIdx.x * blockDim.x + threadIdx.x;
    if (i < n * HH) {
        int f = i & 127;
        float rn = 1.0f / (float)n;
        float m  = sum[f] * rn;
        float v  = sq[f] * rn - m * m;
        float y  = (a[i] - m) * rsqrtf(v + EPS) * g[f] + be[f];
        a[i] = fmaxf(y, 0.0f);
    }
}

// global_add_pool
__global__ void pool_add(const float* __restrict__ a, const int* __restrict__ batch,
                         float* __restrict__ p, int n) {
    int i = blockIdx.x * blockDim.x + threadIdx.x;
    if (i < n * HH) {
        int node = i >> 7, f = i & 127;
        unsafeAtomicAdd(&p[(size_t)batch[node] * HH + f], a[i]);
    }
}

// ---------------------------------------------------------------------------
// Stage W (128x128 f32, row-major) into LDS as bf16 WMMA B-fragments laid out
// in per-(kb,jb,lane) contiguous 32-byte chunks:
//   sWf[((kb*8 + jb)*32 + lane)*16 + i]  (ISA 16-bit B layout: col=lane&15,
//   K-half = lane>>4; i 0..7 -> K=kb*32+kh+i, i 8..15 -> K=kb*32+16+kh+(i-8))
// ---------------------------------------------------------------------------
__device__ __forceinline__ void stage_wfrag(const float* __restrict__ W,
                                            __bf16* __restrict__ sWf, int tid) {
    for (int id = tid; id < 1024; id += 256) {
        int kb   = id >> 8;
        int jb   = (id >> 5) & 7;
        int lane = id & 31;
        int kh   = (lane >> 4) * 8;
        int col  = jb * 16 + (lane & 15);
        v16bf f;
#pragma unroll
        for (int i = 0; i < 8; ++i)
            f[i]     = (__bf16)W[(size_t)(kb * 32 + kh + i) * HH + col];
#pragma unroll
        for (int i = 0; i < 8; ++i)
            f[i + 8] = (__bf16)W[(size_t)(kb * 32 + 16 + kh + i) * HH + col];
        *(v16bf*)(sWf + (size_t)id * 16) = f;
    }
}

// Load the 4 A-fragments (16x32 bf16 each) for one 16-row tile from f32 X.
__device__ __forceinline__ void load_afrag(const float* __restrict__ X, int row,
                                           int lane, v16bf af[4]) {
    const int kh = (lane >> 4) * 8;
#pragma unroll
    for (int kb = 0; kb < 4; ++kb) {
        const float* xp = X + (size_t)row * HH + kb * 32 + kh;
#pragma unroll
        for (int i = 0; i < 8; ++i) af[kb][i]     = (__bf16)xp[i];
#pragma unroll
        for (int i = 0; i < 8; ++i) af[kb][i + 8] = (__bf16)xp[16 + i];
    }
}

// ---------------------------------------------------------------------------
// WMMA GEMM:  Y[rows,128] = X[rows,128] @ W[128,128]   (bf16 in, f32 acc)
// One wave per 16x128 output tile.  W tile is pulled into LDS by the Tensor
// Data Mover (TDM) when available, then re-swizzled into fragment order.
// ---------------------------------------------------------------------------
__global__ void __launch_bounds__(256)
gemm128(const float* __restrict__ X, const float* __restrict__ W,
        float* __restrict__ Y, int rows) {
    __shared__ __align__(32) __bf16 sWf[HH * HH];   // 32KB fragment store
#if HAVE_TDM
    __shared__ __align__(16) float  sW32[HH * HH];  // 64KB TDM destination
    if (threadIdx.x == 0) {
        // Build Tensor DMA Descriptor (D#) per CDNA5 ISA ch.8:
        uint64_t ga = (uint64_t)(uintptr_t)W;
        unsigned int lds_off = (unsigned int)(uintptr_t)(void*)sW32;
        v4u g0;
        g0.x = 1u;                                       // count=1 (valid), user mode
        g0.y = lds_off;                                  // lds_addr (bytes)
        g0.z = (unsigned int)(ga & 0xFFFFFFFFull);       // global_addr[31:0]
        g0.w = (unsigned int)((ga >> 32) & 0x1FFFFFFull) // global_addr[56:32]
             | (2u << 30);                               // type = 2 ("image")
        v8i g1;
        g1[0] = (2 << 16);        // workgroup_mask=0, data_size=2 (4-byte elems)
        g1[1] = (HH << 16);       // tensor_dim0[15:0]  (bits 63:48)
        g1[2] = (HH << 16);       // tensor_dim1[15:0]  (bits 95:80)
        g1[3] = (HH << 16);       // tile_dim0          (bits 127:112)
        g1[4] = HH;               // tile_dim1 (bits 143:128), tile_dim2=0
        g1[5] = HH;               // tensor_dim0_stride[31:0]
        g1[6] = 0;
        g1[7] = 0;
        v4i g2 = {0, 0, 0, 0};
        v4i g3 = {0, 0, 0, 0};
#if defined(__clang_major__) && (__clang_major__ >= 23)
        v8i g4 = {0, 0, 0, 0, 0, 0, 0, 0};
        __builtin_amdgcn_tensor_load_to_lds(g0, g1, g2, g3, g4, 0);
#else
        __builtin_amdgcn_tensor_load_to_lds(g0, g1, g2, g3, 0);
#endif
        __builtin_amdgcn_s_wait_tensorcnt(0);
    }
    __syncthreads();
    const float* wsrc = sW32;
#else
    const float* wsrc = W;
#endif

    stage_wfrag(wsrc, sWf, threadIdx.x);
    __syncthreads();

    const int wave = threadIdx.x >> 5;
    const int lane = threadIdx.x & 31;
    const int m0   = (blockIdx.x * 8 + wave) * 16;
    if (m0 >= rows) return;

    v16bf af[4];
    load_afrag(X, m0 + (lane & 15), lane, af);

    const int col   = lane & 15;
    const int rbase = m0 + ((lane >> 4) << 3);
#pragma unroll
    for (int jb = 0; jb < 8; ++jb) {
        v8f acc = {};
#pragma unroll
        for (int kb = 0; kb < 4; ++kb) {
            v16bf b = *(const v16bf*)(sWf + (size_t)((kb * 8 + jb) * 32 + lane) * 16);
            acc = __builtin_amdgcn_wmma_f32_16x16x32_bf16(
                false, af[kb], false, b, (short)0, acc, false, false);
        }
        float* yp = Y + (size_t)rbase * HH + jb * 16 + col;
#pragma unroll
        for (int r = 0; r < 8; ++r) yp[(size_t)r * HH] = acc[r];
    }
}

// ---------------------------------------------------------------------------
// MLP layer 1: Hid[G,128] = relu(P[G,128] @ Wm1 + bm1)   (WMMA, row-clamped)
// ---------------------------------------------------------------------------
__global__ void __launch_bounds__(256)
mlp1(const float* __restrict__ P, const float* __restrict__ Wm1,
     const float* __restrict__ bm1, float* __restrict__ Hid, int rows) {
    __shared__ __align__(32) __bf16 sWf[HH * HH];
    stage_wfrag(Wm1, sWf, threadIdx.x);
    __syncthreads();

    const int wave = threadIdx.x >> 5;
    const int lane = threadIdx.x & 31;
    const int m0   = (blockIdx.x * 8 + wave) * 16;

    int row = m0 + (lane & 15);
    if (row >= rows) row = rows - 1;     // clamp A loads; stores are guarded
    v16bf af[4];
    load_afrag(P, row, lane, af);

    const int col   = lane & 15;
    const int rbase = m0 + ((lane >> 4) << 3);
#pragma unroll
    for (int jb = 0; jb < 8; ++jb) {
        v8f acc = {};
#pragma unroll
        for (int kb = 0; kb < 4; ++kb) {
            v16bf b = *(const v16bf*)(sWf + (size_t)((kb * 8 + jb) * 32 + lane) * 16);
            acc = __builtin_amdgcn_wmma_f32_16x16x32_bf16(
                false, af[kb], false, b, (short)0, acc, false, false);
        }
        const float bia = bm1[jb * 16 + col];
#pragma unroll
        for (int r = 0; r < 8; ++r) {
            int rr = rbase + r;
            if (rr < rows)
                Hid[(size_t)rr * HH + jb * 16 + col] = fmaxf(acc[r] + bia, 0.0f);
        }
    }
}

// out[g] = Hid[g,:] . Wm2 + bm2
__global__ void mlp2(const float* __restrict__ Hid, const float* __restrict__ Wm2,
                     const float* __restrict__ bm2, float* __restrict__ out) {
    __shared__ float red[HH];
    int g = blockIdx.x, t = threadIdx.x;
    red[t] = Hid[(size_t)g * HH + t] * Wm2[t];
    __syncthreads();
    for (int s = 64; s > 0; s >>= 1) {
        if (t < s) red[t] += red[t + s];
        __syncthreads();
    }
    if (t == 0) out[g] = red[0] + bm2[0];
}

// ---------------------------------------------------------------------------
extern "C" void kernel_launch(void* const* d_in, const int* in_sizes, int n_in,
                              void* d_out, int out_size, void* d_ws, size_t ws_size,
                              hipStream_t stream) {
    const float* x     = (const float*)d_in[0];
    const int*   ei    = (const int*)d_in[1];
    const int*   srcI  = ei;
    const int*   dstI  = ei + EE;
    const int*   batch = (const int*)d_in[2];
    const float* W[3]  = {(const float*)d_in[3],  (const float*)d_in[7],  (const float*)d_in[11]};
    const float* b[3]  = {(const float*)d_in[4],  (const float*)d_in[8],  (const float*)d_in[12]};
    const float* ga[3] = {(const float*)d_in[5],  (const float*)d_in[9],  (const float*)d_in[13]};
    const float* be[3] = {(const float*)d_in[6],  (const float*)d_in[10], (const float*)d_in[14]};
    const float* Wm1   = (const float*)d_in[15];
    const float* bm1   = (const float*)d_in[16];
    const float* Wm2   = (const float*)d_in[17];
    const float* bm2   = (const float*)d_in[18];

    // workspace layout (floats)
    float* ws     = (float*)d_ws;
    float* deg    = ws;                   // N
    float* dis    = deg + NN;             // N
    float* inv    = dis + NN;             // N
    float* sum    = inv + NN;             // 128
    float* sq     = sum + HH;             // 128
    float* pooled = sq  + HH;             // G*128
    float* hid    = pooled + GG * HH;     // G*128
    float* Y      = hid    + GG * HH;     // N*128  (gemm out)
    float* A      = Y      + (size_t)NN * HH; // N*128 (conv out / next input)

    const int TB = 256;
    const int nElem  = NN * HH;                    // 6.4M
    const int gElem  = (nElem + TB - 1) / TB;      // 25000
    const int gGemm  = (NN / 16 + 7) / 8;          // 391 blocks, 8 waves each

    // degrees
    zero_f32 <<<(NN + TB - 1) / TB, TB, 0, stream>>>(deg, NN);
    deg_count<<<(EE + TB - 1) / TB, TB, 0, stream>>>(dstI, deg, EE);
    deg_final<<<(NN + TB - 1) / TB, TB, 0, stream>>>(deg, dis, inv, NN);

    const float* in = x;
    for (int l = 0; l < 3; ++l) {
        gemm128     <<<gGemm, TB, 0, stream>>>(in, W[l], Y, NN);
        self_init   <<<gElem, TB, 0, stream>>>(Y, inv, b[l], A, NN);
        edge_scatter<<<EE / 2, TB, 0, stream>>>(Y, srcI, dstI, dis, A, EE);
        zero_f32    <<<1, TB, 0, stream>>>(sum, 2 * HH);   // sum & sq contiguous
        bn_stats    <<<256, TB, 0, stream>>>(A, sum, sq, NN);
        bn_apply    <<<gElem, TB, 0, stream>>>(A, sum, sq, ga[l], be[l], NN);
        in = A;
    }

    zero_f32<<<(GG * HH + TB - 1) / TB, TB, 0, stream>>>(pooled, GG * HH);
    pool_add<<<gElem, TB, 0, stream>>>(A, batch, pooled, NN);
    mlp1    <<<2, TB, 0, stream>>>(pooled, Wm1, bm1, hid, GG);
    mlp2    <<<GG, HH, 0, stream>>>(hid, Wm2, bm2, (float*)d_out);
}
